// GATLayer_25598005084422
// MI455X (gfx1250) — compile-verified, matrix operand order checked by reading, not run on hardware
//
#include <hip/hip_runtime.h>
#include <hip/hip_bf16.h>

#define N_SENT 100000
#define N_TYPE 10000
#define NEDGE  640000
#define DIM    128
#define SLOPE  0.01f

typedef __attribute__((ext_vector_type(16))) _Float16 v16h;
typedef __attribute__((ext_vector_type(8)))  float    v8f;

// ---------------------------------------------------------------------------
// Kernel 1: scores s[r] = dot(H[r,:], w[:]) for 16 rows per wave via WMMA.
// A tile = f32 products (h*w) rounded to f16, laid out per the documented
// 16-bit A-matrix 16x32 layout. B = splat(1.0h) -> layout independent.
// D[m][n] = sum_k A[m][k] for every n; accumulate over 4 K-chunks in f32.
// ---------------------------------------------------------------------------
__global__ void gat_score_wmma(const float* __restrict__ H,
                               const float* __restrict__ wv,
                               float* __restrict__ out, int nrows) {
    const int lane = threadIdx.x & 31;
    const int wave = (blockIdx.x * blockDim.x + threadIdx.x) >> 5;
    const int r0 = wave * 16;
    if (r0 >= nrows) return;                 // wave-uniform guard (EXEC stays all-1)

    const int m    = lane & 15;              // row this lane contributes to A
    const int koff = (lane < 16) ? 0 : 8;    // A layout: low lanes K 0-7/16-23, high 8-15/24-31

    v16h ones;
#pragma unroll
    for (int i = 0; i < 16; ++i) ones[i] = (_Float16)1.0f;

    v8f c = {};
#pragma unroll
    for (int k0 = 0; k0 < DIM; k0 += 32) {
        const float* hp = H + (size_t)(r0 + m) * DIM + k0;
        const float* wp = wv + k0;
        v16h a;
#pragma unroll
        for (int j = 0; j < 8; ++j)
            a[j] = (_Float16)(hp[koff + j] * wp[koff + j]);
#pragma unroll
        for (int j = 0; j < 8; ++j)
            a[8 + j] = (_Float16)(hp[16 + koff + j] * wp[16 + koff + j]);
        c = __builtin_amdgcn_wmma_f32_16x16x32_f16(
                /*neg_a=*/false, a, /*neg_b=*/false, ones,
                /*c_mod=*/(short)0, c, /*reuse_a=*/false, /*reuse_b=*/false);
    }
    // C/D layout: VGPR r holds row r (lanes 0-15) / row 8+r (lanes 16-31), any N.
    if (lane == 0) {
#pragma unroll
        for (int r = 0; r < 8; ++r) out[r0 + r] = c[r];
    }
    if (lane == 16) {
#pragma unroll
        for (int r = 0; r < 8; ++r) out[r0 + 8 + r] = c[r];
    }
}

// ---------------------------------------------------------------------------
// Kernel 2: zero degree + cursor counters.
// ---------------------------------------------------------------------------
__global__ void gat_zero(int* __restrict__ deg, int* __restrict__ cur, int n) {
    int i = blockIdx.x * blockDim.x + threadIdx.x;
    if (i < n) { deg[i] = 0; cur[i] = 0; }
}

// ---------------------------------------------------------------------------
// Kernel 3: per-destination degree histogram.
// ---------------------------------------------------------------------------
__global__ void gat_degree(const int* __restrict__ dst, int* __restrict__ deg, int e) {
    int i = blockIdx.x * blockDim.x + threadIdx.x;
    if (i < e) atomicAdd(&deg[dst[i]], 1);
}

// ---------------------------------------------------------------------------
// Kernel 4: single-block exclusive scan of deg[0..n) -> row_start[0..n].
// ---------------------------------------------------------------------------
__global__ void gat_scan(const int* __restrict__ deg, int* __restrict__ row_start, int n) {
    __shared__ int sh[1024];
    const int t = threadIdx.x;
    int carry = 0;
    for (int base = 0; base < n; base += 1024) {
        int v = (base + t < n) ? deg[base + t] : 0;
        sh[t] = v;
        __syncthreads();
        for (int off = 1; off < 1024; off <<= 1) {
            int y = (t >= off) ? sh[t - off] : 0;
            __syncthreads();
            sh[t] += y;
            __syncthreads();
        }
        int incl  = sh[t];
        int total = sh[1023];
        if (base + t < n) row_start[base + t] = carry + incl - v;
        __syncthreads();
        carry += total;
    }
    if (t == 0) row_start[n] = carry;
}

// ---------------------------------------------------------------------------
// Kernel 5: CSR fill — bucket edge ids by destination.
// ---------------------------------------------------------------------------
__global__ void gat_fill(const int* __restrict__ dst, const int* __restrict__ row_start,
                         int* __restrict__ cur, int* __restrict__ eidx, int e) {
    int i = blockIdx.x * blockDim.x + threadIdx.x;
    if (i < e) {
        int d = dst[i];
        int pos = atomicAdd(&cur[d], 1);
        eidx[row_start[d] + pos] = i;
    }
}

// ---------------------------------------------------------------------------
// Kernel 6: one wave32 per destination: segment softmax + weighted gather-sum.
// ---------------------------------------------------------------------------
__global__ void gat_aggregate(const float* __restrict__ h_sent,
                              const float* __restrict__ h_type,
                              const float* __restrict__ s_src,
                              const float* __restrict__ s_dst,
                              const int*   __restrict__ srcI,
                              const int*   __restrict__ row_start,
                              const int*   __restrict__ eidx,
                              float*       __restrict__ ws_ex,
                              float*       __restrict__ out) {
    const int wid  = (blockIdx.x * blockDim.x + threadIdx.x) >> 5;   // dst node
    const int lane = threadIdx.x & 31;
    if (wid >= N_TYPE) return;

    const int start = row_start[wid];
    const int end   = row_start[wid + 1];

    float4* orow = (float4*)(out + (size_t)wid * DIM);
    if (end == start) {                 // isolated node: keep h_type
        const float4* trow = (const float4*)(h_type + (size_t)wid * DIM);
        orow[lane] = trow[lane];
        return;
    }
    const float sdt = s_dst[wid];

    // pass 1: segment max of leaky_relu scores
    float mx = -3.402823466e38f;
    for (int i = start + lane; i < end; i += 32) {
        float v = s_src[srcI[eidx[i]]] + sdt;
        v = (v > 0.0f) ? v : SLOPE * v;
        mx = fmaxf(mx, v);
    }
#pragma unroll
    for (int off = 16; off > 0; off >>= 1) mx = fmaxf(mx, __shfl_xor(mx, off, 32));

    // pass 2: exp and segment sum (cache exp values in CSR order)
    float sm = 0.0f;
    for (int i = start + lane; i < end; i += 32) {
        float v = s_src[srcI[eidx[i]]] + sdt;
        v = (v > 0.0f) ? v : SLOPE * v;
        float ex = __expf(v - mx);
        ws_ex[i] = ex;
        sm += ex;
    }
#pragma unroll
    for (int off = 16; off > 0; off >>= 1) sm += __shfl_xor(sm, off, 32);
    const float inv = 1.0f / sm;

    // pass 3: acc += alpha * h_sent[src]; lanes parallel over 128 features
    float4 acc = {0.0f, 0.0f, 0.0f, 0.0f};
    for (int i = start; i < end; ++i) {
        if (i + 1 < end)  // pull next gathered row toward the caches
            __builtin_prefetch(h_sent + (size_t)srcI[eidx[i + 1]] * DIM + lane * 4, 0, 3);
        const float a = ws_ex[i] * inv;
        const float4 hv = *(const float4*)(h_sent + (size_t)srcI[eidx[i]] * DIM + lane * 4);
        acc.x += a * hv.x; acc.y += a * hv.y; acc.z += a * hv.z; acc.w += a * hv.w;
    }
    orow[lane] = acc;
}

// ---------------------------------------------------------------------------
extern "C" void kernel_launch(void* const* d_in, const int* in_sizes, int n_in,
                              void* d_out, int out_size, void* d_ws, size_t ws_size,
                              hipStream_t stream) {
    const float* h_sent = (const float*)d_in[0];
    const float* h_type = (const float*)d_in[1];
    const float* attn_w = (const float*)d_in[2];   // [256]: w_src | w_dst
    const int*   srcI   = (const int*)d_in[3];
    const int*   dstI   = (const int*)d_in[4];
    float*       out    = (float*)d_out;

    // workspace carve-up (16B aligned slices)
    char* ws = (char*)d_ws;
    float* s_src     = (float*)(ws);                         // 400000 B
    float* s_dst     = (float*)(ws + 400000);                //  40000 B
    int*   deg       = (int*)  (ws + 440000);                //  40000 B
    int*   cursor    = (int*)  (ws + 480000);                //  40000 B
    int*   row_start = (int*)  (ws + 520000);                //  40016 B
    int*   eidx      = (int*)  (ws + 560016);                // 2560000 B
    float* ws_ex     = (float*)(ws + 3120016);               // 2560000 B
    (void)ws_size; (void)n_in; (void)in_sizes; (void)out_size;

    // 1-2) dense scores via WMMA (16 rows per wave, 4 waves per block)
    {
        int waves = N_SENT / 16;                              // 6250
        int blocks = (waves + 3) / 4;
        gat_score_wmma<<<blocks, 128, 0, stream>>>(h_sent, attn_w, s_src, N_SENT);
    }
    {
        int waves = N_TYPE / 16;                              // 625
        int blocks = (waves + 3) / 4;
        gat_score_wmma<<<blocks, 128, 0, stream>>>(h_type, attn_w + DIM, s_dst, N_TYPE);
    }

    // 3) zero counters
    gat_zero<<<(N_TYPE + 255) / 256, 256, 0, stream>>>(deg, cursor, N_TYPE);

    // 4) degree histogram
    gat_degree<<<(NEDGE + 255) / 256, 256, 0, stream>>>(dstI, deg, NEDGE);

    // 5) exclusive scan -> row_start
    gat_scan<<<1, 1024, 0, stream>>>(deg, row_start, N_TYPE);

    // 6) CSR fill
    gat_fill<<<(NEDGE + 255) / 256, 256, 0, stream>>>(dstI, row_start, cursor, eidx, NEDGE);

    // 7) per-destination softmax + weighted feature aggregation
    {
        int blocks = (N_TYPE * 32 + 255) / 256;               // 8 waves per block
        gat_aggregate<<<blocks, 256, 0, stream>>>(h_sent, h_type, s_src, s_dst,
                                                  srcI, row_start, eidx, ws_ex, out);
    }
}
